// PsychometricMoE_57037165691564
// MI455X (gfx1250) — matrix operand order repaired and checked by reading, not compiled, not run on hardware
//
#include <hip/hip_runtime.h>
#include <hip/hip_bf16.h>

#define BB   2048
#define FF   512
#define NHH  256
#define DD   1024
#define HHID 2048
#define EE   8
#define MAXROWS 2304          // BB + EE*32 padding, multiple of 32
#define MAXTILES (MAXROWS/32) // 72 tiles of 32 rows

typedef __attribute__((ext_vector_type(16))) _Float16 v16h;
typedef __attribute__((ext_vector_type(8)))  _Float16 v8h;
typedef __attribute__((ext_vector_type(8)))  float    v8f;

// ---------------------------------------------------------------------------
// WMMA fragment loaders (gfx1250 V_WMMA_F32_16X16X32_F16 layouts, ISA 7.12.2)
// ---------------------------------------------------------------------------
__device__ inline v16h load_a_frag(const _Float16* __restrict__ A, int lda,
                                   int m0, int k0) {
  int lane = threadIdx.x & 31;
  int row  = m0 + (lane & 15);
  int c0   = (lane < 16) ? 0 : 8;
  const _Float16* p = A + (size_t)row * lda + k0 + c0;
  v8h lo = *(const v8h*)(p);
  v8h hi = *(const v8h*)(p + 16);
  return __builtin_shufflevector(lo, hi, 0,1,2,3,4,5,6,7,8,9,10,11,12,13,14,15);
}

__device__ inline v16h load_b_frag(const _Float16* __restrict__ BT, int ldb,
                                   int n0, int k0) {
  int lane = threadIdx.x & 31;
  int col  = n0 + (lane & 15);
  int base = (lane < 16) ? 0 : 8;
  const _Float16* p = BT + (size_t)col * ldb + k0 + base;
  v8h lo = *(const v8h*)(p);
  v8h hi = *(const v8h*)(p + 16);
  return __builtin_shufflevector(lo, hi, 0,1,2,3,4,5,6,7,8,9,10,11,12,13,14,15);
}

#define WMMA(a, b, c) __builtin_amdgcn_wmma_f32_16x16x32_f16(false, (a), false, (b), (short)0, (c), false, false)

// One wave computes a 32(M) x 64(N) tile. No manual pipelining / no unrolling:
// keeps live set ~150 VGPRs (8 accs + 6 fragments) so nothing spills; the 12
// loads of iteration i+1 issue while iteration i's WMMAs run on the XDL pipe.
__device__ inline void gemm32_core(const _Float16* __restrict__ A, int lda,
                                   const _Float16* __restrict__ BT, int ldb,
                                   int K, int m0, int n0, v8f acc[8]) {
#pragma clang loop unroll(disable)
  for (int k0 = 0; k0 < K; k0 += 32) {
    if (k0 + 64 < K) {
      __builtin_prefetch(A  + (size_t)(m0 + (threadIdx.x & 15)) * lda + k0 + 96, 0, 1);
      __builtin_prefetch(BT + (size_t)(n0 + (threadIdx.x & 15)) * ldb + k0 + 96, 0, 1);
    }
    v16h a0 = load_a_frag(A, lda, m0,      k0);
    v16h a1 = load_a_frag(A, lda, m0 + 16, k0);
    v16h b0 = load_b_frag(BT, ldb, n0 +  0, k0);
    v16h b1 = load_b_frag(BT, ldb, n0 + 16, k0);
    v16h b2 = load_b_frag(BT, ldb, n0 + 32, k0);
    v16h b3 = load_b_frag(BT, ldb, n0 + 48, k0);
    acc[0] = WMMA(a0, b0, acc[0]);
    acc[1] = WMMA(a1, b0, acc[1]);
    acc[2] = WMMA(a0, b1, acc[2]);
    acc[3] = WMMA(a1, b1, acc[3]);
    acc[4] = WMMA(a0, b2, acc[4]);
    acc[5] = WMMA(a1, b2, acc[5]);
    acc[6] = WMMA(a0, b3, acc[6]);
    acc[7] = WMMA(a1, b3, acc[7]);
  }
}

// ---------------------------------------------------------------------------
// Conversion kernels
// ---------------------------------------------------------------------------
__global__ void k_cvt_h(const float* __restrict__ s, _Float16* __restrict__ d, long n) {
  long i  = (long)blockIdx.x * blockDim.x + threadIdx.x;
  long st = (long)gridDim.x * blockDim.x;
  for (; i < n; i += st) d[i] = (_Float16)s[i];
}

// [E][K][N] f32 -> [E][N][K] f16 via LDS 32x32 tiles (coalesced both sides).
// grid = (Nd/32, Kd/32, E), block = 256 (= 32x8)
__global__ void k_cvt_t(const float* __restrict__ s, _Float16* __restrict__ d,
                        int Kd, int Nd) {
  __shared__ _Float16 t[32][33];
  int e  = blockIdx.z;
  int kb = blockIdx.y * 32, nb = blockIdx.x * 32;
  int tx = threadIdx.x & 31, ty = threadIdx.x >> 5;
  for (int r = ty; r < 32; r += 8)
    t[r][tx] = (_Float16)s[((size_t)e * Kd + kb + r) * Nd + nb + tx];
  __syncthreads();
  for (int r = ty; r < 32; r += 8)
    d[((size_t)e * Nd + nb + r) * Kd + kb + tx] = t[tx][r];
}

// ---------------------------------------------------------------------------
// Generic GEMM: f32 out, optional bias/relu. grid = (N/64, M/32), block = 32
// ---------------------------------------------------------------------------
__global__ void k_gemm(const _Float16* __restrict__ A, int lda,
                       const _Float16* __restrict__ BT, int ldb, int K,
                       float* __restrict__ C, int ldc,
                       const float* __restrict__ bias, int relu) {
  int m0 = blockIdx.y * 32, n0 = blockIdx.x * 64;
  v8f z = {};
  v8f acc[8] = {z, z, z, z, z, z, z, z};
  gemm32_core(A, lda, BT, ldb, K, m0, n0, acc);
  int lane = threadIdx.x & 31;
  int nlo  = lane & 15;
  int mofs = (lane < 16) ? 0 : 8;
  for (int j = 0; j < 4; ++j) {
    int n = n0 + 16 * j + nlo;
    float bv = bias ? bias[n] : 0.f;
    for (int h = 0; h < 2; ++h) {
      int mb = m0 + 16 * h + mofs;
      for (int r = 0; r < 8; ++r) {
        float v = acc[2 * j + h][r] + bv;
        if (relu) v = fmaxf(v, 0.f);
        C[(size_t)(mb + r) * ldc + n] = v;
      }
    }
  }
}

// Expert GEMM1: g[Mg,D] x e_w1T[e][H,D] -> relu -> h1 f16 [Mg,H]
__global__ void k_gemm_e1(const _Float16* __restrict__ G,
                          const _Float16* __restrict__ W1T,
                          const float* __restrict__ eb1,
                          const int* __restrict__ tileExp,
                          _Float16* __restrict__ H1) {
  int e = tileExp[blockIdx.y];
  if (e < 0) return;                       // uniform per block: EXEC stays full
  int m0 = blockIdx.y * 32, n0 = blockIdx.x * 64;
  const _Float16* BT = W1T + (size_t)e * HHID * DD;
  v8f z = {};
  v8f acc[8] = {z, z, z, z, z, z, z, z};
  gemm32_core(G, DD, BT, DD, DD, m0, n0, acc);
  int lane = threadIdx.x & 31;
  int nlo  = lane & 15;
  int mofs = (lane < 16) ? 0 : 8;
  for (int j = 0; j < 4; ++j) {
    int n = n0 + 16 * j + nlo;
    float bv = eb1[e * HHID + n];
    for (int h = 0; h < 2; ++h) {
      int mb = m0 + 16 * h + mofs;
      for (int r = 0; r < 8; ++r) {
        float v = fmaxf(acc[2 * j + h][r] + bv, 0.f);
        H1[(size_t)(mb + r) * HHID + n] = (_Float16)v;
      }
    }
  }
}

// Expert GEMM2: h1[Mg,H] x e_w2T[e][D,H] + b2, scatter to refined by token id
__global__ void k_gemm_e2(const _Float16* __restrict__ H1,
                          const _Float16* __restrict__ W2T,
                          const float* __restrict__ eb2,
                          const int* __restrict__ tileExp,
                          const int* __restrict__ perm,
                          _Float16* __restrict__ refinedH) {
  int e = tileExp[blockIdx.y];
  if (e < 0) return;
  int m0 = blockIdx.y * 32, n0 = blockIdx.x * 64;
  const _Float16* BT = W2T + (size_t)e * DD * HHID;
  v8f z = {};
  v8f acc[8] = {z, z, z, z, z, z, z, z};
  gemm32_core(H1, HHID, BT, HHID, HHID, m0, n0, acc);
  int lane = threadIdx.x & 31;
  int nlo  = lane & 15;
  int mofs = (lane < 16) ? 0 : 8;
  for (int j = 0; j < 4; ++j) {
    int n = n0 + 16 * j + nlo;
    float bv = eb2[e * DD + n];
    for (int h = 0; h < 2; ++h) {
      int mb = m0 + 16 * h + mofs;
      for (int r = 0; r < 8; ++r) {
        int tok = perm[mb + r];
        if (tok >= 0) refinedH[(size_t)tok * DD + n] = (_Float16)(acc[2 * j + h][r] + bv);
      }
    }
  }
}

// ---------------------------------------------------------------------------
// LayerNorm: block per row; row staged in LDS via async-load-to-LDS (ASYNCcnt)
// y = (x+bias - m)/sqrt(v+eps)*g + beta [+relu]
// ---------------------------------------------------------------------------
__global__ void k_ln(const float* __restrict__ X, const float* __restrict__ bias,
                     const float* __restrict__ g, const float* __restrict__ beta,
                     float* __restrict__ outF, _Float16* __restrict__ outH,
                     int N, int relu) {
  __shared__ float xs[DD];
  __shared__ float s1[256], s2[256];
  int row = blockIdx.x;
  const float* x = X + (size_t)row * N;
  // async copy row -> LDS, 16B per lane (N/4 chunks, N <= 1024)
  if ((int)threadIdx.x < (N >> 2)) {
    unsigned int       ldsoff = (unsigned int)(uintptr_t)(&xs[threadIdx.x * 4]);
    unsigned long long gaddr  = (unsigned long long)(uintptr_t)(x + threadIdx.x * 4);
    asm volatile("global_load_async_to_lds_b128 %0, %1, off"
                 :: "v"(ldsoff), "v"(gaddr) : "memory");
  }
  asm volatile("s_wait_asynccnt 0x0" ::: "memory");
  __syncthreads();

  float a = 0.f, b = 0.f;
  for (int c = threadIdx.x; c < N; c += 256) {
    float v = xs[c] + (bias ? bias[c] : 0.f);
    a += v; b += v * v;
  }
  s1[threadIdx.x] = a; s2[threadIdx.x] = b;
  __syncthreads();
  for (int st = 128; st > 0; st >>= 1) {
    if ((int)threadIdx.x < st) { s1[threadIdx.x] += s1[threadIdx.x + st]; s2[threadIdx.x] += s2[threadIdx.x + st]; }
    __syncthreads();
  }
  float m   = s1[0] / N;
  float var = s2[0] / N - m * m;
  float inv = rsqrtf(var + 1e-5f);
  for (int c = threadIdx.x; c < N; c += 256) {
    float v = (xs[c] + (bias ? bias[c] : 0.f) - m) * inv * g[c] + beta[c];
    if (relu) v = fmaxf(v, 0.f);
    if (outF) outF[(size_t)row * N + c] = v;
    if (outH) outH[(size_t)row * N + c] = (_Float16)v;
  }
}

// ---------------------------------------------------------------------------
// Router
// ---------------------------------------------------------------------------
__global__ void k_logits(const float* __restrict__ hr, const float* __restrict__ w2,
                         const float* __restrict__ b2, float* __restrict__ logits) {
  int t = blockIdx.x * 256 + threadIdx.x;        // t over B*8
  if (t >= BB * EE) return;
  int b = t >> 3, e = t & 7;
  const float* row = hr + (size_t)b * 512;
  float s = b2[e];
  for (int k = 0; k < 512; ++k) s += row[k] * w2[k * EE + e];
  logits[t] = s;
}

__global__ void k_softmax(const float* __restrict__ logits, float* __restrict__ weights,
                          float* __restrict__ entbuf, int* __restrict__ top1,
                          int* __restrict__ count) {
  int b = blockIdx.x * 256 + threadIdx.x;
  if (b >= BB) return;
  float l[EE], mx = -1e30f;
  for (int e = 0; e < EE; ++e) { l[e] = logits[b * EE + e]; mx = fmaxf(mx, l[e]); }
  float s = 0.f;
  for (int e = 0; e < EE; ++e) { l[e] = expf(l[e] - mx); s += l[e]; }
  float inv = 1.f / s, ent = 0.f, best = -1.f;
  int am = 0;
  for (int e = 0; e < EE; ++e) {
    float w = l[e] * inv;
    weights[b * EE + e] = w;
    ent -= w * logf(w + 1e-12f);
    if (w > best) { best = w; am = e; }      // first max, matches argmax
  }
  entbuf[b] = ent;
  top1[b]   = am;
  atomicAdd(&count[am], 1);
}

__global__ void k_init(int* count, int* cursor) {
  if (threadIdx.x < EE) { count[threadIdx.x] = 0; cursor[threadIdx.x] = 0; }
}

// single block: init perm/tileExp, 32-padded prefix sum over expert counts
__global__ void k_offsets(const int* __restrict__ count, int* __restrict__ offArr,
                          int* __restrict__ cursor, int* __restrict__ perm,
                          int* __restrict__ tileExp) {
  for (int i = threadIdx.x; i < MAXROWS; i += 256) perm[i] = -1;
  for (int i = threadIdx.x; i < MAXTILES; i += 256) tileExp[i] = -1;
  if (threadIdx.x < EE) cursor[threadIdx.x] = 0;
  __syncthreads();
  if (threadIdx.x == 0) {
    int off = 0;
    for (int e = 0; e < EE; ++e) {
      offArr[e] = off;
      int pc = (count[e] + 31) & ~31;
      for (int t = off / 32; t < (off + pc) / 32; ++t) tileExp[t] = e;
      off += pc;
    }
  }
}

__global__ void k_scatter(const int* __restrict__ top1, const int* __restrict__ offArr,
                          int* __restrict__ cursor, int* __restrict__ perm) {
  int b = blockIdx.x * 256 + threadIdx.x;
  if (b >= BB) return;
  int e   = top1[b];
  int pos = atomicAdd(&cursor[e], 1);
  perm[offArr[e] + pos] = b;
}

__global__ void k_gather(const int* __restrict__ perm, const _Float16* __restrict__ fusedH,
                         _Float16* __restrict__ G) {
  int r = blockIdx.x;
  int d = blockIdx.y * 256 + threadIdx.x;
  int t = perm[r];
  G[(size_t)r * DD + d] = (t >= 0) ? fusedH[(size_t)t * DD + d] : (_Float16)0.f;
}

// ---------------------------------------------------------------------------
// Heads final layers + scalar stats
// ---------------------------------------------------------------------------
__global__ void k_heads_out(const float* __restrict__ Ht, const float* __restrict__ Hi,
                            const float* __restrict__ Hc,
                            const float* __restrict__ tw2, const float* __restrict__ tb2,
                            const float* __restrict__ iw2, const float* __restrict__ ib2,
                            const float* __restrict__ cw2, const float* __restrict__ cb2,
                            float* __restrict__ out) {
  int b = blockIdx.x, o = threadIdx.x;
  if (o >= 18) return;
  const float *hh, *w2, *b2; float* dst; int no, oo;
  if (o < 5)       { hh = Ht; w2 = tw2; b2 = tb2; no = 5;  oo = o;     dst = out + (size_t)b * 5 + oo; }
  else if (o < 8)  { hh = Hi; w2 = iw2; b2 = ib2; no = 3;  oo = o - 5; dst = out + 10240 + (size_t)b * 3 + oo; }
  else             { hh = Hc; w2 = cw2; b2 = cb2; no = 10; oo = o - 8; dst = out + 16384 + (size_t)b * 10 + oo; }
  const float* row = hh + (size_t)b * 256;
  float s = b2[oo];
  for (int k = 0; k < 256; ++k) s += row[k] * w2[k * no + oo];
  *dst = s;
}

__global__ void k_reduce(const float* __restrict__ weights, const float* __restrict__ entbuf,
                         float* __restrict__ out) {
  __shared__ float sred[256 * 9];
  float acc[9];
  for (int j = 0; j < 9; ++j) acc[j] = 0.f;
  for (int t = threadIdx.x; t < BB; t += 256) {
    for (int e = 0; e < EE; ++e) acc[e] += weights[t * EE + e];
    acc[8] += entbuf[t];
  }
  for (int j = 0; j < 9; ++j) sred[threadIdx.x * 9 + j] = acc[j];
  __syncthreads();
  for (int st = 128; st > 0; st >>= 1) {
    if ((int)threadIdx.x < st)
      for (int j = 0; j < 9; ++j) sred[threadIdx.x * 9 + j] += sred[(threadIdx.x + st) * 9 + j];
    __syncthreads();
  }
  if (threadIdx.x == 0) {
    float lb = 0.f;
    for (int e = 0; e < EE; ++e) { float d = sred[e] / BB - 1.f / EE; lb += d * d; }
    out[36864] = lb / EE;
    out[36865] = sred[8] / BB;
  }
}

// ---------------------------------------------------------------------------
extern "C" void kernel_launch(void* const* d_in, const int* in_sizes, int n_in,
                              void* d_out, int out_size, void* d_ws, size_t ws_size,
                              hipStream_t stream) {
  const float* x      = (const float*)d_in[0];
  const float* np_w   = (const float*)d_in[1];
  const float* np_b   = (const float*)d_in[2];
  const float* np_lg  = (const float*)d_in[3];
  const float* np_lb  = (const float*)d_in[4];
  const float* fus_wn = (const float*)d_in[5];
  const float* fus_b  = (const float*)d_in[6];
  const float* fus_lg = (const float*)d_in[7];
  const float* fus_lb = (const float*)d_in[8];
  const float* r_lg   = (const float*)d_in[9];
  const float* r_lb   = (const float*)d_in[10];
  const float* r_w1   = (const float*)d_in[11];
  const float* r_b1   = (const float*)d_in[12];
  const float* r_w2   = (const float*)d_in[13];
  const float* r_b2   = (const float*)d_in[14];
  const float* e_w1   = (const float*)d_in[15];
  const float* e_b1   = (const float*)d_in[16];
  const float* e_w2   = (const float*)d_in[17];
  const float* e_b2   = (const float*)d_in[18];
  const float* th_w1  = (const float*)d_in[19];
  const float* th_b1  = (const float*)d_in[20];
  const float* th_w2  = (const float*)d_in[21];
  const float* th_b2  = (const float*)d_in[22];
  const float* ih_w1  = (const float*)d_in[23];
  const float* ih_b1  = (const float*)d_in[24];
  const float* ih_w2  = (const float*)d_in[25];
  const float* ih_b2  = (const float*)d_in[26];
  const float* ch_w1  = (const float*)d_in[27];
  const float* ch_b1  = (const float*)d_in[28];
  const float* ch_w2  = (const float*)d_in[29];
  const float* ch_b2  = (const float*)d_in[30];
  float* out = (float*)d_out;

  size_t cur = 0;
  auto alloc = [&](size_t bytes) -> void* {
    cur = (cur + 255) & ~(size_t)255;
    void* p = (char*)d_ws + cur;
    cur += bytes;
    return p;
  };
  _Float16* xh      = (_Float16*)alloc((size_t)BB * FF * 2);
  _Float16* npw_t   = (_Float16*)alloc((size_t)NHH * FF * 2);
  _Float16* fusw_t  = (_Float16*)alloc((size_t)DD * NHH * 2);
  _Float16* rw1_t   = (_Float16*)alloc((size_t)512 * DD * 2);
  _Float16* ew1_t   = (_Float16*)alloc((size_t)EE * HHID * DD * 2);
  _Float16* ew2_t   = (_Float16*)alloc((size_t)EE * DD * HHID * 2);
  _Float16* thw1_t  = (_Float16*)alloc((size_t)256 * DD * 2);
  _Float16* ihw1_t  = (_Float16*)alloc((size_t)256 * DD * 2);
  _Float16* chw1_t  = (_Float16*)alloc((size_t)256 * DD * 2);
  float*    tmp0    = (float*)alloc((size_t)BB * NHH * 4);
  _Float16* numembH = (_Float16*)alloc((size_t)BB * NHH * 2);
  float*    tmp1    = (float*)alloc((size_t)BB * DD * 4);
  float*    fused   = (float*)alloc((size_t)BB * DD * 4);
  _Float16* fusedH  = (_Float16*)alloc((size_t)BB * DD * 2);
  _Float16* rlnH    = (_Float16*)alloc((size_t)BB * DD * 2);
  float*    hrout   = (float*)alloc((size_t)BB * 512 * 4);
  float*    logits  = (float*)alloc((size_t)BB * EE * 4);
  float*    weights = (float*)alloc((size_t)BB * EE * 4);
  float*    entbuf  = (float*)alloc((size_t)BB * 4);
  _Float16* Gh      = (_Float16*)alloc((size_t)MAXROWS * DD * 2);
  _Float16* H1h     = (_Float16*)alloc((size_t)MAXROWS * HHID * 2);
  _Float16* refH    = (_Float16*)alloc((size_t)BB * DD * 2);
  float*    HhT     = (float*)alloc((size_t)BB * 256 * 4);
  float*    HhI     = (float*)alloc((size_t)BB * 256 * 4);
  float*    HhC     = (float*)alloc((size_t)BB * 256 * 4);
  int*      top1    = (int*)alloc((size_t)BB * 4);
  int*      perm    = (int*)alloc((size_t)MAXROWS * 4);
  int*      tileExp = (int*)alloc((size_t)MAXTILES * 4);
  int*      count   = (int*)alloc(EE * 4);
  int*      cursor  = (int*)alloc(EE * 4);
  int*      offArr  = (int*)alloc(EE * 4);

  hipLaunchKernelGGL(k_init, dim3(1), dim3(32), 0, stream, count, cursor);

  // f16 conversions (weights transposed to [N][K], LDS-tiled)
  hipLaunchKernelGGL(k_cvt_h, dim3(4096), dim3(256), 0, stream, x, xh, (long)BB * FF);
  hipLaunchKernelGGL(k_cvt_t, dim3(NHH / 32,  FF / 32,   1),  dim3(256), 0, stream, np_w,   npw_t,  FF,   NHH);
  hipLaunchKernelGGL(k_cvt_t, dim3(DD / 32,   NHH / 32,  1),  dim3(256), 0, stream, fus_wn, fusw_t, NHH,  DD);
  hipLaunchKernelGGL(k_cvt_t, dim3(512 / 32,  DD / 32,   1),  dim3(256), 0, stream, r_w1,   rw1_t,  DD,   512);
  hipLaunchKernelGGL(k_cvt_t, dim3(HHID / 32, DD / 32,   EE), dim3(256), 0, stream, e_w1,   ew1_t,  DD,   HHID);
  hipLaunchKernelGGL(k_cvt_t, dim3(DD / 32,   HHID / 32, EE), dim3(256), 0, stream, e_w2,   ew2_t,  HHID, DD);
  hipLaunchKernelGGL(k_cvt_t, dim3(256 / 32,  DD / 32,   1),  dim3(256), 0, stream, th_w1,  thw1_t, DD,   256);
  hipLaunchKernelGGL(k_cvt_t, dim3(256 / 32,  DD / 32,   1),  dim3(256), 0, stream, ih_w1,  ihw1_t, DD,   256);
  hipLaunchKernelGGL(k_cvt_t, dim3(256 / 32,  DD / 32,   1),  dim3(256), 0, stream, ch_w1,  chw1_t, DD,   256);

  // numeric_processor: GEMM -> (+bias) LN relu
  hipLaunchKernelGGL(k_gemm, dim3(NHH / 64, BB / 32), dim3(32), 0, stream,
                     xh, FF, npw_t, FF, FF, tmp0, NHH, (const float*)nullptr, 0);
  hipLaunchKernelGGL(k_ln, dim3(BB), dim3(256), 0, stream,
                     tmp0, np_b, np_lg, np_lb, (float*)nullptr, numembH, NHH, 1);

  // fusion: GEMM -> (+bias) LN relu -> fused (f32 + f16)
  hipLaunchKernelGGL(k_gemm, dim3(DD / 64, BB / 32), dim3(32), 0, stream,
                     numembH, NHH, fusw_t, NHH, NHH, tmp1, DD, (const float*)nullptr, 0);
  hipLaunchKernelGGL(k_ln, dim3(BB), dim3(256), 0, stream,
                     tmp1, fus_b, fus_lg, fus_lb, fused, fusedH, DD, 1);

  // router: LN(fused) -> GEMM+bias+relu -> logits -> softmax/top1
  hipLaunchKernelGGL(k_ln, dim3(BB), dim3(256), 0, stream,
                     fused, (const float*)nullptr, r_lg, r_lb, (float*)nullptr, rlnH, DD, 0);
  hipLaunchKernelGGL(k_gemm, dim3(512 / 64, BB / 32), dim3(32), 0, stream,
                     rlnH, DD, rw1_t, DD, DD, hrout, 512, r_b1, 1);
  hipLaunchKernelGGL(k_logits, dim3(BB * EE / 256), dim3(256), 0, stream, hrout, r_w2, r_b2, logits);
  hipLaunchKernelGGL(k_softmax, dim3(BB / 256), dim3(256), 0, stream, logits, weights, entbuf, top1, count);

  // bucket tokens by expert (padded to 32 rows per bucket), gather f16 rows
  hipLaunchKernelGGL(k_offsets, dim3(1), dim3(256), 0, stream, count, offArr, cursor, perm, tileExp);
  hipLaunchKernelGGL(k_scatter, dim3(BB / 256), dim3(256), 0, stream, top1, offArr, cursor, perm);
  hipLaunchKernelGGL(k_gather, dim3(MAXROWS, DD / 256), dim3(256), 0, stream, perm, fusedH, Gh);

  // grouped expert FFN (top-1 only), scatter back by token id
  hipLaunchKernelGGL(k_gemm_e1, dim3(HHID / 64, MAXTILES), dim3(32), 0, stream,
                     Gh, ew1_t, e_b1, tileExp, H1h);
  hipLaunchKernelGGL(k_gemm_e2, dim3(DD / 64, MAXTILES), dim3(32), 0, stream,
                     H1h, ew2_t, e_b2, tileExp, perm, refH);

  // heads: hidden GEMMs (WMMA) + tiny output dots
  hipLaunchKernelGGL(k_gemm, dim3(256 / 64, BB / 32), dim3(32), 0, stream,
                     refH, DD, thw1_t, DD, DD, HhT, 256, th_b1, 1);
  hipLaunchKernelGGL(k_gemm, dim3(256 / 64, BB / 32), dim3(32), 0, stream,
                     refH, DD, ihw1_t, DD, DD, HhI, 256, ih_b1, 1);
  hipLaunchKernelGGL(k_gemm, dim3(256 / 64, BB / 32), dim3(32), 0, stream,
                     refH, DD, chw1_t, DD, DD, HhC, 256, ch_b1, 1);
  hipLaunchKernelGGL(k_heads_out, dim3(BB), dim3(32), 0, stream,
                     HhT, HhI, HhC, th_w2, th_b2, ih_w2, ih_b2, ch_w2, ch_b2, out);

  // scalar stats (deterministic tree reduce)
  hipLaunchKernelGGL(k_reduce, dim3(1), dim3(256), 0, stream, weights, entbuf, out);

  (void)in_sizes; (void)n_in; (void)out_size; (void)ws_size;
}
